// UltraAgent_76982993814018
// MI455X (gfx1250) — matrix operand order repaired and checked by reading, not compiled, not run on hardware
//
#include <hip/hip_runtime.h>
#include <hip/hip_bf16.h>
#include <hip/math_functions.h>

typedef __attribute__((ext_vector_type(16))) __bf16 v16bf;
typedef __attribute__((ext_vector_type(8)))  __bf16 bf16x8;
typedef __attribute__((ext_vector_type(8)))  float  v8f;

// Problem sizes (fixed by the reference)
constexpr int E_TOT = 4096;
constexpr int A_TOT = 64;
constexpr int D_IN  = 256;
constexpr int HID   = 512;
constexpr int HEADD = 32;

constexpr int MT = 64;   // env rows per block
constexpr int NP = 64;   // N-panel width for shared layers
constexpr int NTHREADS = 256;

// LDS pitches (elements): +8 bf16 (16B) pad -> bank-group spread, keeps 16B alignment
constexpr int PX = D_IN + 8;   // 264
constexpr int PH = HID + 8;    // 520
constexpr int PW = HID + 8;    // 520 (weight panel pitch, shared by all stages)

constexpr size_t SZ_X  = (size_t)MT * PX * 2;        // 33792
constexpr size_t SZ_WB = (size_t)NP * PW * 2;        // 66560 (one weight panel buffer)
constexpr size_t SZ_W  = 2 * SZ_WB;                  // 133120 (double buffered)
constexpr size_t SZ_H  = (size_t)MT * PH * 2;        // 66560
constexpr size_t SZ_F  = (size_t)MT * PH * 2;        // 66560
constexpr size_t SZ_G  = (size_t)MT * HEADD * 4;     // 8192
constexpr size_t SMEM_BYTES = SZ_X + SZ_W + SZ_H + SZ_F + SZ_G; // 308224 <= 320KB WGP LDS

// Pre-transposed bf16 weights in workspace (elements):
// W1t [512][256], W2t [512][512], HW1t [64][32][512]
constexpr size_t W1T_ELEMS  = (size_t)HID * D_IN;           // 131072
constexpr size_t W2T_ELEMS  = (size_t)HID * HID;            // 262144
constexpr size_t HW1T_ELEMS = (size_t)A_TOT * HEADD * HID;  // 1048576
constexpr size_t W2T_OFF  = W1T_ELEMS;
constexpr size_t HW1T_OFF = W1T_ELEMS + W2T_ELEMS;
constexpr size_t WT_TOTAL = W1T_ELEMS + W2T_ELEMS + HW1T_ELEMS; // 1441792 elems (2.88 MB)

__device__ __forceinline__ __bf16 f2bf(float f) { return (__bf16)f; }

#define WAIT_ASYNC(n) asm volatile("s_wait_asynccnt %0" :: "n"(n) : "memory")

// Async copy of one 16-byte chunk: global (bf16, pre-transposed) -> LDS.
// ASYNCcnt-tracked; LDS address is flat-pointer low 32 bits (ISA: addr[31:0] = LDS addr).
__device__ __forceinline__ void async_copy16(uint32_t lds_addr, const __bf16* gsrc) {
  asm volatile("global_load_async_to_lds_b128 %0, %1, off"
               :: "v"(lds_addr), "v"(gsrc) : "memory");
}

// Issue one weight panel (rows of [n][k] bf16, 16B chunks) into an LDS buffer.
template<int CHUNKS_PER_THREAD, int KCHUNKS>
__device__ __forceinline__ void issue_panel(const __bf16* gbase, int gpitch,
                                            __bf16* lbuf, int tid) {
  #pragma unroll
  for (int i = 0; i < CHUNKS_PER_THREAD; ++i) {
    int c  = tid + i * NTHREADS;
    int nl = c / KCHUNKS;
    int kc = c - nl * KCHUNKS;
    const __bf16* g = gbase + (size_t)nl * gpitch + (size_t)kc * 8;
    uint32_t l = (uint32_t)(uintptr_t)(lbuf + nl * PW + kc * 8);
    async_copy16(l, g);
  }
}

union AFrag { v16bf v; bf16x8 h[2]; };

// A fragment (16x32 bf16) from row-major LDS [m][k]:
// lane l: row m0+(l&15); elems 0..7 <- K=k0+8*(l>>4)+i, elems 8..15 <- +16
__device__ __forceinline__ v16bf load_a(const __bf16* base, int pitch, int m0, int k0, int lane) {
  const __bf16* row = base + (m0 + (lane & 15)) * pitch + k0 + ((lane >> 4) << 3);
  AFrag f;
  f.h[0] = *(const bf16x8*)(row);
  f.h[1] = *(const bf16x8*)(row + 16);
  return f.v;
}

// B fragment (32x16 bf16) from transposed LDS [n][k]:
// lane l: col n0+(l&15); elems 0..7 <- K=k0+8*(l>>4)+i, elems 8..15 <- +16
__device__ __forceinline__ v16bf load_b(const __bf16* base, int pitch, int n0, int k0, int lane) {
  const __bf16* row = base + (n0 + (lane & 15)) * pitch + k0 + ((lane >> 4) << 3);
  AFrag f;
  f.h[0] = *(const bf16x8*)(row);
  f.h[1] = *(const bf16x8*)(row + 16);
  return f.v;
}

// One M-tile x two N-tiles, sharing the A fragment: 3 ds_load_b128 per WMMA.
__device__ __forceinline__ void gemm_tile2(const __bf16* sA, int pa,
                                           const __bf16* sB, int pb,
                                           int m0, int nb, int K, int lane,
                                           v8f& acc0, v8f& acc1) {
  #pragma unroll 4
  for (int k0 = 0; k0 < K; k0 += 32) {
    v16bf a  = load_a(sA, pa, m0, k0, lane);
    v16bf b0 = load_b(sB, pb, nb,      k0, lane);
    v16bf b1 = load_b(sB, pb, nb + 16, k0, lane);
    acc0 = __builtin_amdgcn_wmma_f32_16x16x32_bf16(false, a, false, b0,
                                                   (short)0, acc0, false, false);
    acc1 = __builtin_amdgcn_wmma_f32_16x16x32_bf16(false, a, false, b1,
                                                   (short)0, acc1, false, false);
  }
}

__device__ __forceinline__ v8f gemm_tile(const __bf16* sA, int pa,
                                         const __bf16* sB, int pb,
                                         int m0, int nloc, int K, int lane) {
  v8f acc = {};
  #pragma unroll 4
  for (int k0 = 0; k0 < K; k0 += 32) {
    v16bf a = load_a(sA, pa, m0, k0, lane);
    v16bf b = load_b(sB, pb, nloc, k0, lane);
    acc = __builtin_amdgcn_wmma_f32_16x16x32_bf16(false, a, false, b,
                                                  (short)0, acc, false, false);
  }
  return acc;
}

// D-fragment (16x16 f32): lane l holds col n=(l&15), rows m0 + r + 8*(l>>4)
__device__ __forceinline__ void store_relu_bf16(__bf16* dst, int pitch, int m0, int ncol0,
                                                v8f acc, float bias, int lane) {
  int n  = ncol0 + (lane & 15);
  int mo = m0 + ((lane >> 4) << 3);
  #pragma unroll
  for (int r = 0; r < 8; ++r)
    dst[(mo + r) * pitch + n] = f2bf(fmaxf(acc[r] + bias, 0.f));
}

// ---------------- prep: f32 weights -> bf16, pre-transposed [n][k], in workspace --------
extern "C" __global__ __launch_bounds__(256)
void prep_weights(const float* __restrict__ W1, const float* __restrict__ W2,
                  const float* __restrict__ HW1, __bf16* __restrict__ wt) {
  size_t idx = (size_t)blockIdx.x * blockDim.x + threadIdx.x;
  if (idx >= WT_TOTAL) return;
  if (idx < W1T_ELEMS) {                       // W1t[n][k] = W1[k][n]
    int n = (int)(idx >> 8), k = (int)(idx & 255);
    wt[idx] = f2bf(W1[(size_t)k * HID + n]);
  } else if (idx < HW1T_OFF) {                 // W2t[n][k] = W2[k][n]
    size_t j = idx - W2T_OFF;
    int n = (int)(j >> 9), k = (int)(j & 511);
    wt[idx] = f2bf(W2[(size_t)k * HID + n]);
  } else {                                     // HW1t[a][n][k] = HW1[a][k][n]
    size_t j = idx - HW1T_OFF;
    int ag = (int)(j >> 14);
    int r  = (int)(j & 16383);
    int n  = r >> 9, k = r & 511;
    wt[idx] = f2bf(HW1[(size_t)ag * (HID * HEADD) + (size_t)k * HEADD + n]);
  }
}

// ---------------- fused 4-layer MLP, one (agent, 64-env tile) per block ----------------
extern "C" __global__ __launch_bounds__(256)
void fused_agent_mlp(const float* __restrict__ obs,
                     const float* __restrict__ b1, const float* __restrict__ b2,
                     const float* __restrict__ Hb1,
                     const float* __restrict__ HW2, const float* __restrict__ Hb2,
                     const __bf16* __restrict__ wt,
                     float* __restrict__ out)
{
  extern __shared__ char smem[];
  __bf16* sX  = (__bf16*)(smem);
  __bf16* sW0 = (__bf16*)(smem + SZ_X);
  __bf16* sW1b= (__bf16*)(smem + SZ_X + SZ_WB);
  __bf16* sH  = (__bf16*)(smem + SZ_X + SZ_W);
  __bf16* sF  = (__bf16*)(smem + SZ_X + SZ_W + SZ_H);
  float*  sG  = (float*)(smem + SZ_X + SZ_W + SZ_H + SZ_F);

  const int tid  = threadIdx.x;
  const int lane = tid & 31;
  const int wave = tid >> 5;
  const int e0   = blockIdx.x * MT;
  const int ag   = blockIdx.y;

  const __bf16* wtW1 = wt;
  const __bf16* wtW2 = wt + W2T_OFF;
  const __bf16* wtH  = wt + HW1T_OFF + (size_t)ag * HEADD * HID;

  // Per-wave tile assignment (stages 1&2): 4x4 tile grid, one m-row pair of n-tiles / wave
  const int m0 = (wave >> 1) * 16;
  const int nb = (wave & 1) * 32;

  // Kick off stage-1 panel 0 (async, overlaps with obs staging below)
  issue_panel<8, 32>(wtW1, D_IN, sW0, tid);

  // ---------- stage 0: obs tile -> sX (bf16, row-major [m][k]) ----------
  {
    int m = tid >> 2, q = tid & 3;
    const float* src = obs + ((size_t)(e0 + m) * A_TOT + ag) * D_IN + q * 64;
    __bf16* dst = sX + m * PX + q * 64;
    #pragma unroll
    for (int i = 0; i < 64; i += 4) {
      float4 v = *(const float4*)(src + i);
      union { __bf16 b[4]; uint2 u; } pk;
      pk.b[0] = f2bf(v.x); pk.b[1] = f2bf(v.y);
      pk.b[2] = f2bf(v.z); pk.b[3] = f2bf(v.w);
      *(uint2*)(dst + i) = pk.u;
    }
  }

  // ---------- stage 1: sH = relu(sX @ W1 + b1), double-buffered async panels ----------
  for (int p = 0; p < HID / NP; ++p) {
    const int n0 = p * NP;
    __bf16* cur = (p & 1) ? sW1b : sW0;
    if (p < HID / NP - 1) {
      issue_panel<8, 32>(wtW1 + (size_t)(n0 + NP) * D_IN, D_IN,
                         ((p + 1) & 1) ? sW1b : sW0, tid);
      WAIT_ASYNC(8);    // in-order: <=8 outstanding -> panel p landed
    } else {
      // last layer-1 panel: prefetch layer-2 panel 0 into the retired buffer (sW0)
      issue_panel<16, 64>(wtW2, HID, sW0, tid);
      WAIT_ASYNC(16);
    }
    __syncthreads();
    v8f acc0 = {}, acc1 = {};
    gemm_tile2(sX, PX, cur, PW, m0, nb, D_IN, lane, acc0, acc1);
    store_relu_bf16(sH, PH, m0, n0 + nb,      acc0, b1[n0 + nb      + (lane & 15)], lane);
    store_relu_bf16(sH, PH, m0, n0 + nb + 16, acc1, b1[n0 + nb + 16 + (lane & 15)], lane);
    __syncthreads();
  }

  // ---------- stage 2: sF = relu(sH @ W2 + b2) ----------
  for (int p = 0; p < HID / NP; ++p) {
    const int n0 = p * NP;
    __bf16* cur = (p & 1) ? sW1b : sW0;
    if (p < HID / NP - 1) {
      issue_panel<16, 64>(wtW2 + (size_t)(n0 + NP) * HID, HID,
                          ((p + 1) & 1) ? sW1b : sW0, tid);
      WAIT_ASYNC(16);
    } else {
      // last layer-2 panel: prefetch the per-agent head panel into sW0
      issue_panel<8, 64>(wtH, HID, sW0, tid);    // 32 rows x 64 chunks
      WAIT_ASYNC(8);
    }
    __syncthreads();
    v8f acc0 = {}, acc1 = {};
    gemm_tile2(sH, PH, cur, PW, m0, nb, HID, lane, acc0, acc1);
    store_relu_bf16(sF, PH, m0, n0 + nb,      acc0, b2[n0 + nb      + (lane & 15)], lane);
    store_relu_bf16(sF, PH, m0, n0 + nb + 16, acc1, b2[n0 + nb + 16 + (lane & 15)], lane);
    __syncthreads();
  }

  // ---------- stage 3: sG = relu(sF @ HW1[ag] + Hb1[ag])  (64x512x32) ----------
  WAIT_ASYNC(0);     // head panel (issued during stage-2 tail) fully landed
  __syncthreads();
  {
    int hm0 = (wave >> 1) * 16;       // 8 tiles: 4 (m) x 2 (n), one per wave
    int hnl = (wave & 1) * 16;
    v8f acc = gemm_tile(sF, PH, sW0, PW, hm0, hnl, HID, lane);
    float bias = Hb1[ag * HEADD + hnl + (lane & 15)];
    int n  = hnl + (lane & 15);
    int mo = hm0 + ((lane >> 4) << 3);
    #pragma unroll
    for (int r = 0; r < 8; ++r)
      sG[(mo + r) * HEADD + n] = fmaxf(acc[r] + bias, 0.f);
  }
  __syncthreads();

  // ---------- stage 4: out[e, ag] = sigmoid(sG . HW2[ag] + Hb2[ag]) ----------
  if (tid < MT) {
    const float* hw2 = HW2 + ag * HEADD;
    float acc = Hb2[ag];
    #pragma unroll
    for (int k = 0; k < HEADD; ++k)
      acc += sG[tid * HEADD + k] * hw2[k];
    out[(size_t)(e0 + tid) * A_TOT + ag] = 1.f / (1.f + __expf(-acc));
  }
}

extern "C" void kernel_launch(void* const* d_in, const int* in_sizes, int n_in,
                              void* d_out, int out_size, void* d_ws, size_t ws_size,
                              hipStream_t stream) {
  (void)in_sizes; (void)n_in; (void)out_size; (void)ws_size;
  const float* obs = (const float*)d_in[0];
  const float* W1  = (const float*)d_in[1];
  const float* b1  = (const float*)d_in[2];
  const float* W2  = (const float*)d_in[3];
  const float* b2  = (const float*)d_in[4];
  const float* HW1 = (const float*)d_in[5];
  const float* Hb1 = (const float*)d_in[6];
  const float* HW2 = (const float*)d_in[7];
  const float* Hb2 = (const float*)d_in[8];
  float*  out = (float*)d_out;
  __bf16* wt  = (__bf16*)d_ws;   // 2.88 MB of workspace for pre-transposed bf16 weights

  // weights -> bf16 [n][k] (runs every call; deterministic)
  prep_weights<<<(unsigned)((WT_TOTAL + 255) / 256), 256, 0, stream>>>(W1, W2, HW1, wt);

  // >64KB dynamic LDS needs explicit opt-in (host-side, capture-safe, deterministic)
  hipFuncSetAttribute((const void*)fused_agent_mlp,
                      hipFuncAttributeMaxDynamicSharedMemorySize, (int)SMEM_BYTES);

  dim3 grid(E_TOT / MT, A_TOT);
  fused_agent_mlp<<<grid, 256, SMEM_BYTES, stream>>>(obs, b1, b2, Hb1, HW2, Hb2, wt, out);
}
